// GlobalLevelEncoding_90159953477843
// MI455X (gfx1250) — matrix operand order repaired and checked by reading, not compiled, not run on hardware
//
#include <hip/hip_runtime.h>

typedef __attribute__((ext_vector_type(16))) _Float16 v16h;
typedef __attribute__((ext_vector_type(8)))  float    v8f;

#define B_   16
#define N_   128
#define H_   32
#define D_   5
#define NEF_ 3
#define NP1  129

__device__ __forceinline__ float4 f4add(float4 a, float4 b) {
    return make_float4(a.x + b.x, a.y + b.y, a.z + b.z, a.w + b.w);
}

// Pre-swizzle Wdis (D,H,H) f32 -> f16 WMMA B fragments.
// Fragment n = d*2+tile; per lane 16 halves; element e of lane L holds
// Wdis[d, h=(L>=16?8:0)+(e&7)+(e>=8?16:0), k=tile*16+(L&15)]  (B layout: K
// table order matching A, column = lane&15).
__global__ void grax_swizzle_wdis(const float* __restrict__ wdis,
                                  _Float16* __restrict__ wsH) {
    int n = blockIdx.x * blockDim.x + threadIdx.x;
    if (n >= D_ * 2 * 32 * 16) return;
    int e    = n & 15;
    int lane = (n >> 4) & 31;
    int tile = (n >> 9) & 1;
    int d    = n >> 10;
    int h = ((lane >= 16) ? 8 : 0) + (e & 7) + ((e >= 8) ? 16 : 0);
    int k = tile * 16 + (lane & 15);
    wsH[n] = (_Float16)wdis[(d * H_ + h) * H_ + k];
}

// Borders: out[b,h,0,j] for all j, out[b,h,i,0] for i>=1.
__global__ void grax_border(const float* __restrict__ ab,
                            const float* __restrict__ virt,
                            float* __restrict__ out) {
    int tid = blockIdx.x * blockDim.x + threadIdx.x;
    if (tid >= B_ * H_ * 257) return;
    int local = tid % 257;
    int bh = tid / 257;
    int b = bh / H_, h = bh % H_;
    float t = virt[h];
    if (local < NP1) {
        int j = local;
        out[((b * H_ + h) * NP1 + 0) * NP1 + j] =
            2.0f * ab[(b * NP1 + 0) * NP1 + j] + t;
    } else {
        int i = local - NP1 + 1;   // 1..128
        out[((b * H_ + h) * NP1 + i) * NP1 + 0] =
            2.0f * ab[(b * NP1 + i) * NP1 + 0] + t;
    }
}

// One workgroup per (b,i); 8 waves; wave w handles j in [16w, 16w+16).
// GEMM rows = j-pairs, K = d*32+h (5 chunks of 32), N = 32 output channels.
__global__ void __launch_bounds__(256)
grax_main(const float* __restrict__ ab,
          const int*   __restrict__ spatial_pos,
          const int*   __restrict__ edge_input,
          const float* __restrict__ enc_w,
          const float* __restrict__ struct_w,
          const _Float16* __restrict__ wsH,
          float* __restrict__ out) {
    const int lane = threadIdx.x & 31;
    const int wave = threadIdx.x >> 5;
    const int b  = blockIdx.x >> 7;     // /128
    const int i  = blockIdx.x & 127;
    const int j0 = wave * 16;
    const int hi = lane >> 4;           // 0 or 1
    const int lm = lane & 15;

    // B fragments (pre-swizzled): 10 x v16h, 32B-aligned per lane.
    v16h Bf[10];
#pragma unroll
    for (int n = 0; n < 10; ++n)
        Bf[n] = *(const v16h*)(wsH + (n * 32 + lane) * 16);

    v8f acc0 = {};
    v8f acc1 = {};

    // A-matrix row for this lane (16-bit A layout: M = lane&15).
    const int rowj = j0 + lm;
    const int* eidx = edge_input + (((b * N_ + i) * N_ + rowj) * D_) * NEF_;
    const int h0 = hi * 8;   // lanes 0-15: K base 0 ; lanes 16-31: K base 8
    const float third = 1.0f / 3.0f;

#pragma unroll
    for (int d = 0; d < D_; ++d) {
        float4 a0 = make_float4(0.f, 0.f, 0.f, 0.f);
        float4 a1 = a0, a2 = a0, a3 = a0;
#pragma unroll
        for (int f = 0; f < NEF_; ++f) {
            int idx = eidx[d * NEF_ + f];
            const float* p = enc_w + idx * H_ + h0;    // 16B aligned
            a0 = f4add(a0, *(const float4*)(p));       // h0+0..3   (e 0..3)
            a1 = f4add(a1, *(const float4*)(p + 4));   // h0+4..7   (e 4..7)
            a2 = f4add(a2, *(const float4*)(p + 16));  // h0+16..19 (e 8..11)
            a3 = f4add(a3, *(const float4*)(p + 20));  // h0+20..23 (e 12..15)
        }
        v16h A;
        A[0]  = (_Float16)(a0.x * third); A[1]  = (_Float16)(a0.y * third);
        A[2]  = (_Float16)(a0.z * third); A[3]  = (_Float16)(a0.w * third);
        A[4]  = (_Float16)(a1.x * third); A[5]  = (_Float16)(a1.y * third);
        A[6]  = (_Float16)(a1.z * third); A[7]  = (_Float16)(a1.w * third);
        A[8]  = (_Float16)(a2.x * third); A[9]  = (_Float16)(a2.y * third);
        A[10] = (_Float16)(a2.z * third); A[11] = (_Float16)(a2.w * third);
        A[12] = (_Float16)(a3.x * third); A[13] = (_Float16)(a3.y * third);
        A[14] = (_Float16)(a3.z * third); A[15] = (_Float16)(a3.w * third);

        acc0 = __builtin_amdgcn_wmma_f32_16x16x32_f16(
                   false, A, false, Bf[2 * d + 0], (short)0, acc0, false, false);
        acc1 = __builtin_amdgcn_wmma_f32_16x16x32_f16(
                   false, A, false, Bf[2 * d + 1], (short)0, acc1, false, false);
    }

    // Epilogue. C/D layout: VGPR q -> row m = q + 8*hi, col = lm (+16).
    const int iO = i + 1;
#pragma unroll
    for (int q = 0; q < 8; ++q) {
        int j = j0 + hi * 8 + q;
        int sraw = spatial_pos[(b * N_ + i) * N_ + j];
        int spv  = (sraw == 0) ? 1 : ((sraw > 1) ? (sraw - 1) : sraw);
        spv = (spv > D_) ? D_ : spv;
        float rsp = 1.0f / (float)spv;
        float abv = 2.0f * ab[(b * NP1 + iO) * NP1 + (j + 1)];
        float s0 = struct_w[sraw * H_ + lm];
        float s1 = struct_w[sraw * H_ + lm + 16];
        out[((b * H_ + lm)      * NP1 + iO) * NP1 + (j + 1)] = abv + s0 + acc0[q] * rsp;
        out[((b * H_ + lm + 16) * NP1 + iO) * NP1 + (j + 1)] = abv + s1 + acc1[q] * rsp;
    }
}

extern "C" void kernel_launch(void* const* d_in, const int* in_sizes, int n_in,
                              void* d_out, int out_size, void* d_ws, size_t ws_size,
                              hipStream_t stream) {
    // setup_inputs order: x, attn_bias, spatial_pos, edge_input,
    //                     edge_encoder_w, structural_w, edge_dis_w, virt_w
    const float* attn_bias   = (const float*)d_in[1];
    const int*   spatial_pos = (const int*)d_in[2];
    const int*   edge_input  = (const int*)d_in[3];
    const float* enc_w       = (const float*)d_in[4];
    const float* struct_w    = (const float*)d_in[5];
    const float* wdis        = (const float*)d_in[6];
    const float* virt        = (const float*)d_in[7];
    float* out = (float*)d_out;
    _Float16* wsH = (_Float16*)d_ws;   // needs 10 KB

    grax_swizzle_wdis<<<(D_ * 2 * 32 * 16 + 255) / 256, 256, 0, stream>>>(wdis, wsH);
    grax_border<<<(B_ * H_ * 257 + 255) / 256, 256, 0, stream>>>(attn_bias, virt, out);
    grax_main<<<B_ * N_, 256, 0, stream>>>(attn_bias, spatial_pos, edge_input,
                                           enc_w, struct_w, wsH, out);
}
